// ScratchxLSTM_NNS_53412213293471
// MI455X (gfx1250) — compile-verified, matrix operand order checked by reading, not run on hardware
//
#include <hip/hip_runtime.h>
#include <hip/hip_bf16.h>
#include <math.h>
#include <stdint.h>

#define DEV_INLINE __device__ __forceinline__

typedef __bf16 bf16_t;
typedef __attribute__((ext_vector_type(16))) __bf16 v16bf;
typedef __attribute__((ext_vector_type(8)))  __bf16 v8bf;
typedef __attribute__((ext_vector_type(8)))  float  v8f;

constexpr int B_ = 4096, T_ = 64, IN_ = 3, D_ = 128, H_ = 4, NB_ = 2, DH_ = 32;
constexpr float EPS_ = 1e-5f;

DEV_INLINE float warp_sum(float v) {
  #pragma unroll
  for (int m = 16; m; m >>= 1) v += __shfl_xor(v, m, 32);
  return v;
}

DEV_INLINE v8f wmma_bf16(v16bf a, v16bf b, v8f c) {
  return __builtin_amdgcn_wmma_f32_16x16x32_bf16(false, a, false, b, (short)0, c, false, false);
}

// Build a 16x32 bf16 A-fragment from a row-major bf16 row pointer.
// Lane L holds row M=L&15; element e holds K = (e/8)*16 + (L>>4)*8 + (e&7):
// two contiguous 8-element (16B) chunks per 32-wide K tile.
DEV_INLINE v16bf load_afrag(const bf16_t* __restrict__ row, int kt, int g) {
  const v8bf lo = *(const v8bf*)(row + kt * 32 +      g * 8);
  const v8bf hi = *(const v8bf*)(row + kt * 32 + 16 + g * 8);
  v16bf a;
  #pragma unroll
  for (int e = 0; e < 8; ++e) { a[e] = lo[e]; a[8 + e] = hi[e]; }
  return a;
}

// ---------------- input projection: h = x @ w_in + b_in ----------------------------------
__global__ void k_in_proj(const float* __restrict__ x, const float* __restrict__ w,
                          const float* __restrict__ b, float* __restrict__ h) {
  int idx = blockIdx.x * blockDim.x + threadIdx.x;
  if (idx >= B_ * T_ * D_) return;
  int d  = idx & (D_ - 1);
  int bt = idx >> 7;
  const float* xr = x + (size_t)bt * IN_;
  h[idx] = fmaf(xr[0], w[d], fmaf(xr[1], w[D_ + d], fmaf(xr[2], w[2 * D_ + d], b[d])));
}

// ---------------- pack fp32 KxN row-major weight into bf16 WMMA B-fragment order ----------
__global__ void k_pack_bfrag(const float* __restrict__ src, bf16_t* __restrict__ dst,
                             int K, int N, int nbatch, size_t sstride, size_t dstride) {
  int KT = K >> 5, NT = N >> 4;
  size_t total = (size_t)nbatch * NT * KT * 512;
  size_t idx = (size_t)blockIdx.x * blockDim.x + threadIdx.x;
  if (idx >= total) return;
  int e    = (int)(idx & 15);
  int lane = (int)((idx >> 4) & 31);
  size_t f = idx >> 9;
  int kt = (int)(f % KT); f /= KT;
  int nt = (int)(f % NT);
  int bt = (int)(f / NT);
  int n = nt * 16 + (lane & 15);
  int k = kt * 32 + ((e >> 3) << 4) + ((lane >> 4) << 3) + (e & 7);
  dst[(size_t)bt * dstride + (((size_t)nt * KT + kt) << 9) + (lane << 4) + e] =
      (bf16_t)src[(size_t)bt * sstride + (size_t)k * N + n];
}

// ---------------- LayerNorm -> bf16 row-major (wave per row, shuffle reduce) --------------
__global__ __launch_bounds__(256) void k_ln_bf16(const float* __restrict__ h,
                                                 const float* __restrict__ g,
                                                 const float* __restrict__ bb,
                                                 bf16_t* __restrict__ out, int rows) {
  int wid  = (blockIdx.x * blockDim.x + threadIdx.x) >> 5;
  int lane = threadIdx.x & 31;
  if (wid >= rows) return;
  const float* r = h + (size_t)wid * D_;
  float v[4], s = 0.f, sq = 0.f;
  #pragma unroll
  for (int i = 0; i < 4; ++i) { v[i] = r[i * 32 + lane]; s += v[i]; sq += v[i] * v[i]; }
  s = warp_sum(s); sq = warp_sum(sq);
  float mu  = s * (1.f / D_);
  float inv = rsqrtf(sq * (1.f / D_) - mu * mu + EPS_);
  bf16_t* o = out + (size_t)wid * D_;
  #pragma unroll
  for (int i = 0; i < 4; ++i) {
    int cidx = i * 32 + lane;
    o[cidx] = (bf16_t)((v[i] - mu) * inv * g[cidx] + bb[cidx]);
  }
}

// ---------------- fused sLSTM scan ---------------------------------------------------------
// 2 waves per block, 16 batch rows per block. Wave w owns d-tiles jj = 4w..4w+3 (heads 2w,
// 2w+1). c/n/m state in VGPRs (D-frag layout); h_new staged via LDS (f32 for groupnorm,
// bf16 row-major for next-step A-frags). Weight fragments are streamed from L2 every step:
// an opaque zero offset (asm barrier) keeps the loads loop-variant (no hoist -> no spill)
// while the typed kernarg pointer keeps global address-space inference (global_load, not
// flat_load, so the WMMA feed only pays LOADcnt and never couples with the LDS DScnt).
__global__ __launch_bounds__(64) void k_scan(
    const bf16_t* __restrict__ xn, const bf16_t* __restrict__ Wg_pk,
    const float*  __restrict__ bg, const bf16_t* __restrict__ R_pk,
    const float*  __restrict__ gn_g, const float* __restrict__ gn_b,
    float* __restrict__ h) {
  __shared__ __align__(16) float  hf[16 * D_];
  __shared__ __align__(16) bf16_t hb[16 * D_];
  const int tid  = threadIdx.x;
  const int lane = tid & 31, wv = tid >> 5;      // wave 0/1
  const int m = lane & 15, g = lane >> 4;
  const size_t row0 = (size_t)blockIdx.x * 16;

  float c[4][8], nst[4][8], mst[4][8];
  #pragma unroll
  for (int j = 0; j < 4; ++j)
    #pragma unroll
    for (int r = 0; r < 8; ++r) { c[j][r] = 0.f; nst[j][r] = 0.f; mst[j][r] = 0.f; }
  for (int idx = tid; idx < 16 * D_; idx += 64) hb[idx] = (bf16_t)0.f;
  __syncthreads();

  float bgv[4][4];                               // [gate][local j]
  #pragma unroll
  for (int gg = 0; gg < 4; ++gg)
    #pragma unroll
    for (int j = 0; j < 4; ++j) bgv[gg][j] = bg[gg * D_ + (wv * 4 + j) * 16 + m];

  for (int t = 0; t < T_; ++t) {
    // opaque per-iteration offset: loads can't be hoisted/CSE'd across t, but the
    // pointers remain GEPs off kernargs so they still lower to global_load.
    size_t zoff = 0;
    asm volatile("" : "+s"(zoff));
    const bf16_t* WgB = Wg_pk + zoff;
    const bf16_t* RB  = R_pk  + zoff;

    const bf16_t* xrow = xn + ((row0 + m) * T_ + t) * D_;
    if (t + 1 < T_) __builtin_prefetch(xrow + D_, 0, 1);   // global_prefetch_b8

    v16bf axn[4];
    #pragma unroll
    for (int kt = 0; kt < 4; ++kt) axn[kt] = load_afrag(xrow, kt, g);
    v16bf ah[2];                                           // this wave's 2 heads of h_{t-1}
    #pragma unroll
    for (int hl = 0; hl < 2; ++hl) ah[hl] = load_afrag(hb + m * D_, wv * 2 + hl, g);

    #pragma unroll
    for (int j = 0; j < 4; ++j) {
      const int jj = wv * 4 + j;                 // global d-tile 0..7
      const int hh = jj >> 1, sub = jj & 1, hl = hh & 1;
      // preload all 20 B-fragments for this d-tile (kt-major for 4-way WMMA ILP)
      v16bf bw[4][4];
      #pragma unroll
      for (int gg = 0; gg < 4; ++gg)
        #pragma unroll
        for (int kt = 0; kt < 4; ++kt)
          bw[kt][gg] = *(const v16bf*)(WgB + ((((size_t)(gg * 8 + jj)) * 4 + kt) << 9) + (lane << 4));
      v16bf br[4];
      #pragma unroll
      for (int gg = 0; gg < 4; ++gg)
        br[gg] = *(const v16bf*)(RB + ((((size_t)(gg * 4 + hh)) * 2 + sub) << 9) + (lane << 4));

      v8f acc[4];
      #pragma unroll
      for (int gg = 0; gg < 4; ++gg)
        #pragma unroll
        for (int r = 0; r < 8; ++r) acc[gg][r] = bgv[gg][j];
      #pragma unroll
      for (int kt = 0; kt < 4; ++kt)                       // xn @ Wg, 4 independent chains
        #pragma unroll
        for (int gg = 0; gg < 4; ++gg)
          acc[gg] = wmma_bf16(axn[kt], bw[kt][gg], acc[gg]);
      #pragma unroll
      for (int gg = 0; gg < 4; ++gg)                       // + h_{t-1}[head] @ R[g,head]
        acc[gg] = wmma_bf16(ah[hl], br[gg], acc[gg]);

      #pragma unroll
      for (int r = 0; r < 8; ++r) {                        // stabilized exp gates
        float it = acc[0][r], ft = acc[1][r], zt = acc[2][r], ot = acc[3][r];
        float mn = fmaxf(ft + mst[j][r], it);
        float iv = __expf(it - mn);
        float fv = __expf(ft + mst[j][r] - mn);
        float cn = fv * c[j][r] + iv * tanhf(zt);
        float nn = fv * nst[j][r] + iv;
        float hn = (1.f / (1.f + __expf(-ot))) * cn / nn;
        c[j][r] = cn; nst[j][r] = nn; mst[j][r] = mn;
        hf[(r + 8 * g) * D_ + jj * 16 + m] = hn;           // D-frag -> row-major via LDS
      }
    }
    __syncthreads();
    for (int idx = tid; idx < 16 * D_; idx += 64) hb[idx] = (bf16_t)hf[idx];
    {                                                      // fused groupnorm + residual
      int rw = tid >> 2, hh2 = tid & 3;                    // 64 (row,head) pairs
      const float* hrow = hf + rw * D_ + hh2 * DH_;
      float s = 0.f, sq = 0.f;
      #pragma unroll
      for (int k = 0; k < DH_; ++k) { float xv = hrow[k]; s += xv; sq += xv * xv; }
      float mu  = s * (1.f / DH_);
      float inv = rsqrtf(sq * (1.f / DH_) - mu * mu + EPS_);
      float* hg = h + ((row0 + rw) * T_ + t) * D_ + hh2 * DH_;
      #pragma unroll
      for (int k = 0; k < DH_; ++k) {
        int cc = hh2 * DH_ + k;
        hg[k] += (hrow[k] - mu) * inv * gn_g[cc] + gn_b[cc];
      }
    }
    __syncthreads();
  }
}

// ---------------- FFN: h += gelu(yn@W1) @ W2, wave per 16 rows, LDS bf16 stage ------------
__global__ __launch_bounds__(128) void k_ffn(const bf16_t* __restrict__ yn,
                                             const bf16_t* __restrict__ W1_pk,
                                             const bf16_t* __restrict__ W2_pk,
                                             float* __restrict__ h) {
  __shared__ __align__(16) bf16_t ub[4][16 * D_];
  const int lane = threadIdx.x & 31, wv = threadIdx.x >> 5;
  const int m = lane & 15, g = lane >> 4;
  const size_t row0 = ((size_t)blockIdx.x * 4 + wv) * 16;
  v16bf a[4];
  #pragma unroll
  for (int kt = 0; kt < 4; ++kt) a[kt] = load_afrag(yn + (row0 + m) * D_, kt, g);
  #pragma unroll
  for (int nt = 0; nt < 8; ++nt) {
    v8f acc;
    #pragma unroll
    for (int r = 0; r < 8; ++r) acc[r] = 0.f;
    #pragma unroll
    for (int kt = 0; kt < 4; ++kt) {
      const v16bf bw = *(const v16bf*)(W1_pk + ((((size_t)nt) * 4 + kt) << 9) + (lane << 4));
      acc = wmma_bf16(a[kt], bw, acc);
    }
    #pragma unroll
    for (int r = 0; r < 8; ++r) {                          // tanh-approx GELU
      float xv = acc[r];
      float u = 0.5f * xv * (1.f + tanhf(0.7978845608f * (xv + 0.044715f * xv * xv * xv)));
      ub[wv][(r + 8 * g) * D_ + nt * 16 + m] = (bf16_t)u;
    }
  }
  __syncthreads();
  v16bf ua[4];
  #pragma unroll
  for (int kt = 0; kt < 4; ++kt) ua[kt] = load_afrag(&ub[wv][m * D_], kt, g);
  #pragma unroll
  for (int nt = 0; nt < 8; ++nt) {
    v8f acc;
    #pragma unroll
    for (int r = 0; r < 8; ++r) acc[r] = 0.f;
    #pragma unroll
    for (int kt = 0; kt < 4; ++kt) {
      const v16bf bw = *(const v16bf*)(W2_pk + ((((size_t)nt) * 4 + kt) << 9) + (lane << 4));
      acc = wmma_bf16(ua[kt], bw, acc);
    }
    #pragma unroll
    for (int r = 0; r < 8; ++r) {
      size_t ri = (row0 + r + 8 * g) * D_ + nt * 16 + m;
      h[ri] += acc[r];
    }
  }
}

// ---------------- final LN + projection to scalar per batch row ---------------------------
__global__ __launch_bounds__(256) void k_final(const float* __restrict__ h,
                                               const float* __restrict__ g,
                                               const float* __restrict__ b,
                                               const float* __restrict__ wo,
                                               const float* __restrict__ bo,
                                               float* __restrict__ out) {
  int wid  = (blockIdx.x * blockDim.x + threadIdx.x) >> 5;
  int lane = threadIdx.x & 31;
  if (wid >= B_) return;
  const float* r = h + ((size_t)wid * T_ + (T_ - 1)) * D_;
  float v[4], s = 0.f, sq = 0.f;
  #pragma unroll
  for (int i = 0; i < 4; ++i) { v[i] = r[i * 32 + lane]; s += v[i]; sq += v[i] * v[i]; }
  s = warp_sum(s); sq = warp_sum(sq);
  float mu  = s * (1.f / D_);
  float inv = rsqrtf(sq * (1.f / D_) - mu * mu + EPS_);
  float dot = 0.f;
  #pragma unroll
  for (int i = 0; i < 4; ++i) {
    int cidx = i * 32 + lane;
    dot += ((v[i] - mu) * inv * g[cidx] + b[cidx]) * wo[cidx];
  }
  dot = warp_sum(dot);
  if (lane == 0) out[wid] = dot + bo[0];
}

extern "C" void kernel_launch(void* const* d_in, const int* in_sizes, int n_in,
                              void* d_out, int out_size, void* d_ws, size_t ws_size,
                              hipStream_t stream) {
  (void)in_sizes; (void)n_in; (void)out_size; (void)ws_size;
  const float* x     = (const float*)d_in[0];
  const float* w_in  = (const float*)d_in[1];
  const float* b_in  = (const float*)d_in[2];
  const float* ln1_g = (const float*)d_in[3];
  const float* ln1_b = (const float*)d_in[4];
  const float* Wg    = (const float*)d_in[5];
  const float* bg    = (const float*)d_in[6];
  const float* Rg    = (const float*)d_in[7];
  const float* gn_g  = (const float*)d_in[8];
  const float* gn_b  = (const float*)d_in[9];
  const float* ln2_g = (const float*)d_in[10];
  const float* ln2_b = (const float*)d_in[11];
  const float* W1    = (const float*)d_in[12];
  const float* W2    = (const float*)d_in[13];
  const float* lnf_g = (const float*)d_in[14];
  const float* lnf_b = (const float*)d_in[15];
  const float* w_out = (const float*)d_in[16];
  const float* b_out = (const float*)d_in[17];
  float* out = (float*)d_out;

  char* ws = (char*)d_ws;
  size_t off = 0;
  auto walloc = [&](size_t bytes) { size_t o = off; off = (off + bytes + 255) & ~(size_t)255; return o; };
  float*  h   = (float*)(ws + walloc((size_t)B_ * T_ * D_ * sizeof(float)));
  bf16_t* xnb = (bf16_t*)(ws + walloc((size_t)B_ * T_ * D_ * sizeof(bf16_t)));  // xn / yn reuse
  bf16_t *WgP[NB_], *RP[NB_], *W1P[NB_], *W2P[NB_];
  for (int blk = 0; blk < NB_; ++blk) {
    WgP[blk] = (bf16_t*)(ws + walloc((size_t)D_ * 4 * D_ * sizeof(bf16_t)));
    RP[blk]  = (bf16_t*)(ws + walloc((size_t)4 * H_ * DH_ * DH_ * sizeof(bf16_t)));
    W1P[blk] = (bf16_t*)(ws + walloc((size_t)D_ * D_ * sizeof(bf16_t)));
    W2P[blk] = (bf16_t*)(ws + walloc((size_t)D_ * D_ * sizeof(bf16_t)));
  }

  k_in_proj<<<(B_ * T_ * D_ + 255) / 256, 256, 0, stream>>>(x, w_in, b_in, h);

  for (int blk = 0; blk < NB_; ++blk) {
    { size_t tot = (size_t)(512 / 16) * (128 / 32) * 512;   // Wg: K=128, N=512
      k_pack_bfrag<<<(tot + 255) / 256, 256, 0, stream>>>(Wg + (size_t)blk * D_ * 4 * D_, WgP[blk],
                                                          128, 512, 1, 0, 0); }
    { size_t tot = (size_t)16 * 2 * 1 * 512;                // R: 16 matrices of K=32, N=32
      k_pack_bfrag<<<(tot + 255) / 256, 256, 0, stream>>>(Rg + (size_t)blk * 16384, RP[blk],
                                                          32, 32, 16, 1024, 1024); }
    { size_t tot = (size_t)(128 / 16) * (128 / 32) * 512;   // W1/W2: K=128, N=128
      k_pack_bfrag<<<(tot + 255) / 256, 256, 0, stream>>>(W1 + (size_t)blk * 16384, W1P[blk],
                                                          128, 128, 1, 0, 0);
      k_pack_bfrag<<<(tot + 255) / 256, 256, 0, stream>>>(W2 + (size_t)blk * 16384, W2P[blk],
                                                          128, 128, 1, 0, 0); }
  }

  const int rows = B_ * T_;
  for (int blk = 0; blk < NB_; ++blk) {
    k_ln_bf16<<<rows / 8, 256, 0, stream>>>(h, ln1_g + blk * D_, ln1_b + blk * D_, xnb, rows);
    k_scan<<<B_ / 16, 64, 0, stream>>>(xnb, WgP[blk], bg + blk * 4 * D_, RP[blk],
                                       gn_g + blk * D_, gn_b + blk * D_, h);
    k_ln_bf16<<<rows / 8, 256, 0, stream>>>(h, ln2_g + blk * D_, ln2_b + blk * D_, xnb, rows);
    k_ffn<<<rows / 64, 128, 0, stream>>>(xnb, W1P[blk], W2P[blk], h);
  }
  k_final<<<B_ / 8, 256, 0, stream>>>(h, lnf_g, lnf_b, w_out, b_out, out);
}
